// SelfAttention_42752104465020
// MI455X (gfx1250) — matrix operand order repaired and checked by reading, not compiled, not run on hardware
//
#include <hip/hip_runtime.h>
#include <hip/hip_bf16.h>
#include <stdint.h>

// ---------------------------------------------------------------------------
// Self-attention block (B=16, C=512, N=1024, 8 heads x d=64) for gfx1250.
// GEMMs via v_wmma_f32_16x16x32_f16; tile staging via ASYNCcnt-tracked
// global_load_async_to_lds_b128 with double buffering (CDNA5 path).
// ---------------------------------------------------------------------------

typedef __attribute__((ext_vector_type(16))) _Float16 v16h;
typedef __attribute__((ext_vector_type(8)))  _Float16 v8h;
typedef __attribute__((ext_vector_type(8)))  float    v8f;

#define BATCH   16
#define CHN     512
#define NTOK    1024
#define NHEAD   8
#define HDIM    64
#define MTOT    (BATCH * NTOK)   // 16384 rows
#define TM      128              // block-level M tile for GEMMs
#define KSTEP   32

// ---- CDNA5 async copy global -> LDS (ASYNCcnt) -----------------------------
__device__ __forceinline__ void async_ld16(void* lds, const void* gsrc) {
  // low 32 bits of a generic LDS pointer are the LDS byte address (ISA 10.2)
  uint32_t l = (uint32_t)(uintptr_t)lds;
  asm volatile("global_load_async_to_lds_b128 %0, %1, off"
               :: "v"(l), "v"(gsrc) : "memory");
}
__device__ __forceinline__ void wait_async0() {
  asm volatile("s_wait_asynccnt 0x0" ::: "memory");
}

// ---- WMMA fragment helpers (layouts per CDNA5 ISA 7.12.2, wave32) ----------

// A-fragment: 16x32 f16, row-major source (ld halves per row).
// element e: K = ((e>>3)<<4) + ((lane>>4)<<3) + (e&7), row = lane&15
__device__ __forceinline__ v16h load_a_frag(const _Float16* base, int ld) {
  const int lane = threadIdx.x & 31;
  const int row  = lane & 15;
  const int kb   = (lane >> 4) << 3;          // 0 or 8
  const _Float16* p = base + row * ld + kb;
  v8h lo = *(const v8h*)(p);                  // K = kb .. kb+7
  v8h hi = *(const v8h*)(p + 16);             // K = kb+16 .. kb+23
  v16h r;
#pragma unroll
  for (int i = 0; i < 8; ++i) { r[i] = lo[i]; r[8 + i] = hi[i]; }
  return r;
}

// B-fragment: 32x16 f16 from (Nout rows x K cols) row-major ("B^T, K-contig").
// element e: K = ((e>>3)<<4) + ((lane>>4)<<3) + (e&7), column N = lane&15
__device__ __forceinline__ v16h load_b_frag(const _Float16* base, int ld) {
  const int lane = threadIdx.x & 31;
  const int n    = lane & 15;
  const int kb   = (lane >> 4) << 3;          // 0 or 8
  const _Float16* p = base + n * ld + kb;
  v8h lo = *(const v8h*)(p);
  v8h hi = *(const v8h*)(p + 16);
  v16h r;
#pragma unroll
  for (int i = 0; i < 8; ++i) { r[i] = lo[i]; r[8 + i] = hi[i]; }
  return r;
}

__device__ __forceinline__ v8f wmma_f16(v16h a, v16h b, v8f c) {
  return __builtin_amdgcn_wmma_f32_16x16x32_f16(false, a, false, b,
                                                (short)0, c, false, false);
}

// C/D fragment: VGPR r holds row (r + 8*(lane>=16)), column lane&15.

// ---------------------------------------------------------------------------
// K0: LayerNorm over C per token + f16 cast. x (B,C,N); Xn (B*N, C) f16.
// ---------------------------------------------------------------------------
__global__ void __launch_bounds__(256)
k_ln(const float* __restrict__ x, const float* __restrict__ g,
     const float* __restrict__ be, _Float16* __restrict__ Xn) {
  const int t = blockIdx.x * 256 + threadIdx.x;       // 0 .. 16383
  const int b = t >> 10, n = t & 1023;
  const float* xp = x + (size_t)b * CHN * NTOK + n;
  float s = 0.f, s2 = 0.f;
  for (int c = 0; c < CHN; ++c) {
    float v = xp[(size_t)c * NTOK];
    s += v; s2 += v * v;
  }
  const float mu  = s * (1.0f / CHN);
  const float var = s2 * (1.0f / CHN) - mu * mu;
  const float rs  = rsqrtf(var + 1e-5f);
  _Float16* op = Xn + (size_t)t * CHN;
  for (int c = 0; c < CHN; ++c) {
    float v = (xp[(size_t)c * NTOK] - mu) * rs * g[c] + be[c];
    op[c] = (_Float16)v;
  }
}

__global__ void __launch_bounds__(256)
k_cvt(const float* __restrict__ src, _Float16* __restrict__ dst, int n) {
  int i = blockIdx.x * 256 + threadIdx.x;
  if (i < n) dst[i] = (_Float16)src[i];
}

// ---------------------------------------------------------------------------
// Cooperative staging of one k-step: A tile (128x32) + B tile (64x32) -> LDS.
// 256 threads x 3 async b128 copies; ASYNCcnt tracks completion.
// ---------------------------------------------------------------------------
__device__ __forceinline__ void stage_tiles(const _Float16* __restrict__ Ak,
                                            const _Float16* __restrict__ Wk,
                                            _Float16 (* __restrict__ At)[KSTEP],
                                            _Float16 (* __restrict__ Bt)[KSTEP],
                                            int t) {
  {  // B: 64 rows * 4 chunks = 256
    const int row = t >> 2, c = (t & 3) * 8;
    async_ld16(&Bt[row][c], Wk + (size_t)row * CHN + c);
  }
#pragma unroll
  for (int q = 0; q < 2; ++q) {  // A: 128 rows * 4 chunks = 512
    const int id = t + q * 256;
    const int row = id >> 2, c = (id & 3) * 8;
    async_ld16(&At[row][c], Ak + (size_t)row * CHN + c);
  }
}

// ---------------------------------------------------------------------------
// K1: QKV GEMM  Y(16384x1536) = Xn @ qkv_w^T, block computes 128x64 tile.
// Double-buffered async LDS staging; WMMA operands fed from LDS.
// Epilogue scatters Q,K -> (bh,token,d); V -> transposed (bh,d,token).
// ---------------------------------------------------------------------------
__global__ void __launch_bounds__(256)
k_qkv(const _Float16* __restrict__ Xn, const _Float16* __restrict__ Wq,
      _Float16* __restrict__ Qb, _Float16* __restrict__ Kb,
      _Float16* __restrict__ Vt) {
  __shared__ _Float16 At[2][TM][KSTEP];   // 2 x 8 KB
  __shared__ _Float16 Bt[2][64][KSTEP];   // 2 x 4 KB
  const int t = threadIdx.x;
  const int lane = t & 31, wslot = t >> 5;
  const int nb = blockIdx.x % 24;          // 24 column blocks of 64 -> 1536
  const int mblk = blockIdx.x / 24;        // 128 row blocks of 128
  const _Float16* Abase = Xn + (size_t)mblk * TM * CHN;
  const _Float16* Wbase = Wq + (size_t)nb * 64 * CHN;

  stage_tiles(Abase, Wbase, At[0], Bt[0], t);
  v8f acc[4] = {};
#pragma unroll 1
  for (int i = 0; i < CHN / KSTEP; ++i) {
    wait_async0();
    __syncthreads();                       // tile i visible to all waves
    if (i + 1 < CHN / KSTEP)               // DMA next tile during compute
      stage_tiles(Abase + (i + 1) * KSTEP, Wbase + (i + 1) * KSTEP,
                  At[(i + 1) & 1], Bt[(i + 1) & 1], t);
    const int buf = i & 1;
    v16h a = load_a_frag(&At[buf][wslot * 16][0], KSTEP);
#pragma unroll
    for (int j = 0; j < 4; ++j) {
      v16h w = load_b_frag(&Bt[buf][j * 16][0], KSTEP);
      acc[j] = wmma_f16(a, w, acc[j]);
    }
  }
  const int hi = (lane >> 4) << 3;
  const int c16 = lane & 15;
#pragma unroll
  for (int j = 0; j < 4; ++j) {
    const int o     = nb * 64 + j * 16 + c16;   // 0..1535
    const int which = o >> 9;                   // 0=q 1=k 2=v
    const int oc    = o & 511;
    const int h = oc >> 6, d = oc & 63;
#pragma unroll
    for (int r = 0; r < 8; ++r) {
      const int row = mblk * TM + wslot * 16 + r + hi;   // 0..16383
      const int bb = row >> 10, n = row & 1023;
      const size_t bh = (size_t)(bb * NHEAD + h);
      const _Float16 val = (_Float16)acc[j][r];
      if (which == 0)      Qb[(bh * NTOK + n) * HDIM + d] = val;
      else if (which == 1) Kb[(bh * NTOK + n) * HDIM + d] = val;
      else                 Vt[(bh * HDIM + d) * NTOK + n] = val;
    }
  }
}

// ---------------------------------------------------------------------------
// K2: Flash attention, one wave per (b,h,16-row tile); 32-token column blocks.
// Pipelined: V loads issued before S-WMMAs; next K block prefetched before
// softmax; P transposed C->A layout via per-wave LDS tile.
// ---------------------------------------------------------------------------
__global__ void __launch_bounds__(256)
k_attn(const _Float16* __restrict__ Qb, const _Float16* __restrict__ Kb,
       const _Float16* __restrict__ Vt, _Float16* __restrict__ Aout) {
  __shared__ _Float16 Pt[8][16][32];            // per-wave P tile (8 KB)
  const int lane  = threadIdx.x & 31;
  const int wslot = threadIdx.x >> 5;
  const int wave  = blockIdx.x * 8 + wslot;
  const int mb = wave & 63;                     // 64 row tiles per (b,h)
  const int bh = wave >> 6;                     // 0..127
  const _Float16* Qp  = Qb + ((size_t)bh * NTOK + mb * 16) * HDIM;
  const _Float16* Kp0 = Kb + (size_t)bh * NTOK * HDIM;
  const _Float16* Vp0 = Vt + (size_t)bh * HDIM * NTOK;

  const v16h aq0 = load_a_frag(Qp, HDIM);        // d = 0..31
  const v16h aq1 = load_a_frag(Qp + 32, HDIM);   // d = 32..63
  v8f o0 = {}, o1 = {}, o2 = {}, o3 = {};
  float mrow[8], lrow[8];
#pragma unroll
  for (int r = 0; r < 8; ++r) { mrow[r] = -1e30f; lrow[r] = 0.f; }
  const float scale = 0.125f;                    // 1/sqrt(64)
  const int hi = (lane >> 4) << 3;
  const int c16 = lane & 15;

  // K fragments for j0 = 0 (prologue of the register pipeline)
  v16h kb00 = load_b_frag(Kp0, HDIM);
  v16h kb01 = load_b_frag(Kp0 + 32, HDIM);
  v16h kb10 = load_b_frag(Kp0 + 16 * HDIM, HDIM);
  v16h kb11 = load_b_frag(Kp0 + 16 * HDIM + 32, HDIM);

#pragma unroll 1
  for (int j0 = 0; j0 < NTOK; j0 += 32) {
    // V loads issued early: independent of softmax, in flight during S+softmax
    v16h v0 = load_b_frag(Vp0 + j0, NTOK);
    v16h v1 = load_b_frag(Vp0 + 16 * NTOK + j0, NTOK);
    v16h v2 = load_b_frag(Vp0 + 32 * NTOK + j0, NTOK);
    v16h v3 = load_b_frag(Vp0 + 48 * NTOK + j0, NTOK);

    v8f s0 = {}, s1 = {};
    s0 = wmma_f16(aq0, kb00, s0); s0 = wmma_f16(aq1, kb01, s0);
    s1 = wmma_f16(aq0, kb10, s1); s1 = wmma_f16(aq1, kb11, s1);

    // prefetch next block's K fragments while softmax runs
    if (j0 + 32 < NTOK) {
      const _Float16* Kn = Kp0 + (size_t)(j0 + 32) * HDIM;
      kb00 = load_b_frag(Kn, HDIM);
      kb01 = load_b_frag(Kn + 32, HDIM);
      kb10 = load_b_frag(Kn + 16 * HDIM, HDIM);
      kb11 = load_b_frag(Kn + 16 * HDIM + 32, HDIM);
    }

    // online softmax (rows r+8*hi are uniform across each 16-lane half)
#pragma unroll
    for (int r = 0; r < 8; ++r) {
      s0[r] *= scale; s1[r] *= scale;
      float v = fmaxf(s0[r], s1[r]);
      v = fmaxf(v, __shfl_xor(v, 1));
      v = fmaxf(v, __shfl_xor(v, 2));
      v = fmaxf(v, __shfl_xor(v, 4));
      v = fmaxf(v, __shfl_xor(v, 8));
      const float mn = fmaxf(mrow[r], v);
      const float alpha = __expf(mrow[r] - mn);
      s0[r] = __expf(s0[r] - mn);
      s1[r] = __expf(s1[r] - mn);
      float rs = s0[r] + s1[r];
      rs += __shfl_xor(rs, 1); rs += __shfl_xor(rs, 2);
      rs += __shfl_xor(rs, 4); rs += __shfl_xor(rs, 8);
      lrow[r] = lrow[r] * alpha + rs;
      mrow[r] = mn;
      o0[r] *= alpha; o1[r] *= alpha; o2[r] *= alpha; o3[r] *= alpha;
    }
    // P: C-layout -> LDS -> A-layout (wave-private slot, no barrier needed)
#pragma unroll
    for (int r = 0; r < 8; ++r) {
      Pt[wslot][r + hi][c16]      = (_Float16)s0[r];
      Pt[wslot][r + hi][c16 + 16] = (_Float16)s1[r];
    }
    v16h pa = load_a_frag(&Pt[wslot][0][0], 32);
    o0 = wmma_f16(pa, v0, o0); o1 = wmma_f16(pa, v1, o1);
    o2 = wmma_f16(pa, v2, o2); o3 = wmma_f16(pa, v3, o3);
  }
  // normalize and store (b, token, h*64+d) == (B*N, C) row-major
  const int bb = bh >> 3, h = bh & 7;
#pragma unroll
  for (int r = 0; r < 8; ++r) {
    const float inv = 1.0f / lrow[r];
    const int token = mb * 16 + r + hi;
    _Float16* dst = Aout + ((size_t)(bb * NTOK + token)) * CHN + h * HDIM + c16;
    dst[0]  = (_Float16)(o0[r] * inv);
    dst[16] = (_Float16)(o1[r] * inv);
    dst[32] = (_Float16)(o2[r] * inv);
    dst[48] = (_Float16)(o3[r] * inv);
  }
}

// ---------------------------------------------------------------------------
// K3: proj GEMM + bias + residual.  out(B,C,N) = Aout @ proj_w^T + b + x
// Same double-buffered async-LDS structure as k_qkv.
// ---------------------------------------------------------------------------
__global__ void __launch_bounds__(256)
k_proj(const _Float16* __restrict__ Aout, const _Float16* __restrict__ Wp,
       const float* __restrict__ pb, const float* __restrict__ x,
       float* __restrict__ out) {
  __shared__ _Float16 At[2][TM][KSTEP];
  __shared__ _Float16 Bt[2][64][KSTEP];
  const int t = threadIdx.x;
  const int lane = t & 31, wslot = t >> 5;
  const int nb = blockIdx.x & 7;           // 8 column blocks of 64 -> 512
  const int mblk = blockIdx.x >> 3;        // 128 row blocks of 128
  const _Float16* Abase = Aout + (size_t)mblk * TM * CHN;
  const _Float16* Wbase = Wp + (size_t)nb * 64 * CHN;

  stage_tiles(Abase, Wbase, At[0], Bt[0], t);
  v8f acc[4] = {};
#pragma unroll 1
  for (int i = 0; i < CHN / KSTEP; ++i) {
    wait_async0();
    __syncthreads();
    if (i + 1 < CHN / KSTEP)
      stage_tiles(Abase + (i + 1) * KSTEP, Wbase + (i + 1) * KSTEP,
                  At[(i + 1) & 1], Bt[(i + 1) & 1], t);
    const int buf = i & 1;
    v16h a = load_a_frag(&At[buf][wslot * 16][0], KSTEP);
#pragma unroll
    for (int j = 0; j < 4; ++j) {
      v16h w = load_b_frag(&Bt[buf][j * 16][0], KSTEP);
      acc[j] = wmma_f16(a, w, acc[j]);
    }
  }
  const int hi = (lane >> 4) << 3;
  const int c16 = lane & 15;
#pragma unroll
  for (int j = 0; j < 4; ++j) {
    const int cg = nb * 64 + j * 16 + c16;      // output channel 0..511
    const float bias = pb[cg];
#pragma unroll
    for (int r = 0; r < 8; ++r) {
      const int row = mblk * TM + wslot * 16 + r + hi;
      const int bb = row >> 10, n = row & 1023;
      const size_t idx = ((size_t)bb * CHN + cg) * NTOK + n;
      out[idx] = acc[j][r] + bias + x[idx];     // residual in native layout
    }
  }
}

// ---------------------------------------------------------------------------
// Launch
// ---------------------------------------------------------------------------
extern "C" void kernel_launch(void* const* d_in, const int* in_sizes, int n_in,
                              void* d_out, int out_size, void* d_ws, size_t ws_size,
                              hipStream_t stream) {
  const float* x      = (const float*)d_in[0];   // (16, 512, 32, 32)
  const float* qkv_w  = (const float*)d_in[1];   // (1536, 512)
  const float* proj_w = (const float*)d_in[2];   // (512, 512)
  const float* proj_b = (const float*)d_in[3];   // (512,)
  const float* ln_g   = (const float*)d_in[4];   // (512,)
  const float* ln_b   = (const float*)d_in[5];   // (512,)
  float* out = (float*)d_out;                    // (16, 512, 32, 32)

  char* ws = (char*)d_ws;
  const size_t XN_OFF = 0;                       // 16 MB (aliased by Aout)
  const size_t WQ_OFF = 16777216;                // 1.5 MB
  const size_t WP_OFF = WQ_OFF + 1572864;        // 0.5 MB
  const size_t Q_OFF  = WP_OFF + 524288;         // 16 MB
  const size_t K_OFF  = Q_OFF + 16777216;        // 16 MB
  const size_t VT_OFF = K_OFF + 16777216;        // 16 MB  -> total ~66 MB
  _Float16* Xn   = (_Float16*)(ws + XN_OFF);
  _Float16* Aout = (_Float16*)(ws + XN_OFF);     // Xn dead after k_qkv
  _Float16* Wq   = (_Float16*)(ws + WQ_OFF);
  _Float16* Wp   = (_Float16*)(ws + WP_OFF);
  _Float16* Qb   = (_Float16*)(ws + Q_OFF);
  _Float16* Kb   = (_Float16*)(ws + K_OFF);
  _Float16* Vt   = (_Float16*)(ws + VT_OFF);

  k_ln<<<MTOT / 256, 256, 0, stream>>>(x, ln_g, ln_b, Xn);
  k_cvt<<<(3 * CHN * CHN + 255) / 256, 256, 0, stream>>>(qkv_w, Wq, 3 * CHN * CHN);
  k_cvt<<<(CHN * CHN + 255) / 256, 256, 0, stream>>>(proj_w, Wp, CHN * CHN);
  // K1: 128 m-tiles * 24 n-blocks
  k_qkv<<<128 * 24, 256, 0, stream>>>(Xn, Wq, Qb, Kb, Vt);
  // K2: 128 (b,h) * 64 tiles = 8192 waves, 8 per block
  k_attn<<<8192 / 8, 256, 0, stream>>>(Qb, Kb, Vt, Aout);
  // K3: 128 m-tiles * 8 n-blocks
  k_proj<<<128 * 8, 256, 0, stream>>>(Aout, Wp, proj_b, x, out);
}
